// LSTM_12747462934537
// MI455X (gfx1250) — matrix operand order repaired and checked by reading, not compiled, not run on hardware
//
#include <hip/hip_runtime.h>
#include <hip/hip_bf16.h>

typedef unsigned short u16;
typedef unsigned int   u32;

typedef __attribute__((ext_vector_type(16))) __bf16 v16bf;
typedef __attribute__((ext_vector_type(8)))  float  v8f;

#define HIDDEN 128
#define SEQ    28
#define INPUT  28
#define OUTD   10
#define BATCH  16384
#define MTILE  64            // batch rows per workgroup
#define NBLK   (BATCH / MTILE)

// ---- ws layout ----
// swizzled bf16 weight fragment blob (halves):
//   [0      ,16384 )  w_ih_l0   (K padded 28->32, 1 kstep)
//   [16384  ,81920 )  w_hh_l0   (4 ksteps)
//   [81920  ,147456)  w_ih_l1   (4 ksteps)
//   [147456 ,212992)  w_hh_l1   (4 ksteps)
#define WBLOB_HALVES 212992
#define BIAS0_BYTE   425984
#define BIAS1_BYTE   428032
#define H0SEQ_BYTE   430080     // B*T*H bf16 = 117,440,512 bytes

__device__ __forceinline__ u16 f2bf(float f) {
  u32 u = __float_as_uint(f);
  u32 r = u + 0x7FFFu + ((u >> 16) & 1u);
  return (u16)(r >> 16);
}
__device__ __forceinline__ float bf2f(u16 h) { return __uint_as_float(((u32)h) << 16); }
__device__ __forceinline__ float sigf(float v)  { return 1.0f / (1.0f + __expf(-v)); }
// exact identity tanh(x) = 2*sigmoid(2x) - 1, lowers to v_exp/v_rcp TRANS ops
__device__ __forceinline__ float tanhfast(float v) {
  return 2.0f / (1.0f + __expf(-2.0f * v)) - 1.0f;
}

struct Frag32 { uint4 a, b; };

// A-fragment (16x32 bf16, M x K) from row-major bf16 [rows][stride]:
// lane 0-15: M=lane, K = k0..k0+7 (vgpr0-3) and k0+16..k0+23 (vgpr4-7)
// lane16-31: M=lane-16, K offsets +8
__device__ __forceinline__ v16bf loadA(const u16* sA, int stride, int row0, int k0, int lane) {
  const u16* p = sA + (row0 + (lane & 15)) * stride + k0 + ((lane >> 4) << 3);
  Frag32 f;
  f.a = *(const uint4*)p;         // K .. K+7
  f.b = *(const uint4*)(p + 16);  // K+16 .. K+23
  return __builtin_bit_cast(v16bf, f);
}

// B-fragment pre-swizzled: fragment = 512 contiguous halves, lane's 16 halves at lane*16
__device__ __forceinline__ v16bf loadB(const u16* frag, int lane) {
  const u16* p = frag + lane * 16;
  Frag32 f;
  f.a = *(const uint4*)p;
  f.b = *(const uint4*)(p + 8);
  return __builtin_bit_cast(v16bf, f);
}

// -------- weight pre-swizzle + bias fold --------
__global__ __launch_bounds__(256) void prep_kernel(
    const float* __restrict__ wih0, const float* __restrict__ whh0,
    const float* __restrict__ bih0, const float* __restrict__ bhh0,
    const float* __restrict__ wih1, const float* __restrict__ whh1,
    const float* __restrict__ bih1, const float* __restrict__ bhh1,
    u16* __restrict__ wblob, float* __restrict__ bias0, float* __restrict__ bias1) {
  int i = blockIdx.x * blockDim.x + threadIdx.x;
  if (i < WBLOB_HALVES) {
    const float* src; int Ksrc, rem;
    if (i < 16384)        { src = wih0; Ksrc = INPUT;  rem = i; }
    else if (i < 81920)   { src = whh0; Ksrc = HIDDEN; rem = i - 16384; }
    else if (i < 147456)  { src = wih1; Ksrc = HIDDEN; rem = i - 81920; }
    else                  { src = whh1; Ksrc = HIDDEN; rem = i - 147456; }
    int kstep = rem / 16384, r2 = rem % 16384;
    int nt = r2 / 512, r3 = r2 % 512;
    int lane = r3 / 16, e = r3 % 16;
    int k = kstep * 32 + ((lane >> 4) << 3) + (e & 7) + ((e >> 3) << 4);
    int n = lane & 15, gate = nt & 3, hb = nt >> 2;
    int srow = gate * HIDDEN + hb * 16 + n;   // gate-interleaved column remap
    float v = (k < Ksrc) ? src[srow * Ksrc + k] : 0.0f;
    wblob[i] = f2bf(v);
  } else {
    int j = i - WBLOB_HALVES;
    if (j < 1024) {
      int layer = j >> 9, idx = j & 511;
      int nt = idx >> 4, n = idx & 15, gate = nt & 3, hb = nt >> 2;
      int s = gate * HIDDEN + hb * 16 + n;
      if (layer == 0) bias0[idx] = bih0[s] + bhh0[s];
      else            bias1[idx] = bih1[s] + bhh1[s];
    }
  }
}

// -------- fused LSTM layer (WMMA bf16) --------
template <int LAYER>
__global__ __launch_bounds__(256) void lstm_layer_kernel(
    const float* __restrict__ x,        // layer0 only: [B, 784]
    const u16* __restrict__ wih,        // swizzled frag blob
    const u16* __restrict__ whh,
    const float* __restrict__ bias,     // gate-interleaved [512]
    u16* __restrict__ h0seq,            // layer0: write, layer1: read ([B][T][H] bf16)
    const float* __restrict__ fc_w, const float* __restrict__ fc_b,
    float* __restrict__ out) {
  constexpr int KSX     = (LAYER == 0) ? 1 : 4;   // K-steps for input projection
  constexpr int XSTRIDE = (LAYER == 0) ? 32 : HIDDEN;
  constexpr int WIH_H   = (LAYER == 0) ? 16384 : 65536;

  __shared__ alignas(16) u16 sWih[WIH_H];
  __shared__ alignas(16) u16 sWhh[65536];
  __shared__ alignas(16) u16 sX[MTILE * XSTRIDE];
  __shared__ alignas(16) u16 sH[MTILE * HIDDEN];
  __shared__ float sB[512];

  const int tid = threadIdx.x, wave = tid >> 5, lane = tid & 31;
  const int nloc = lane & 15, hiB = lane >> 4;
  const int b0 = blockIdx.x * MTILE;

  // stage weights/bias to LDS, zero h state
  {
    const uint4* g1 = (const uint4*)wih;  uint4* l1 = (uint4*)sWih;
    for (int p = tid; p < WIH_H / 8; p += 256) l1[p] = g1[p];
    const uint4* g2 = (const uint4*)whh;  uint4* l2 = (uint4*)sWhh;
    for (int p = tid; p < 8192; p += 256) l2[p] = g2[p];
    for (int p = tid; p < 512; p += 256) sB[p] = bias[p];
    uint4 z = {0, 0, 0, 0};
    uint4* lh = (uint4*)sH;
    for (int p = tid; p < (MTILE * HIDDEN) / 8; p += 256) lh[p] = z;
  }
  __syncthreads();

  // time-invariant bias fragments: used as the C operand of the first WMMA
  v8f biasv[4];
#pragma unroll
  for (int g = 0; g < 4; ++g) {
    float bb = sB[(wave * 4 + g) * 16 + nloc];
#pragma unroll
    for (int e = 0; e < 8; ++e) biasv[g][e] = bb;
  }

  v8f acc[4][4];   // [m-tile][gate]  -> gates for hidden cols [wave*16, wave*16+16)
  v8f cst[4];      // persistent cell state, same layout as D-matrix tiles
#pragma unroll
  for (int mt = 0; mt < 4; ++mt)
#pragma unroll
    for (int e = 0; e < 8; ++e) cst[mt][e] = 0.0f;

  for (int t = 0; t < SEQ; ++t) {
    // ---- stage this timestep's input into LDS (bf16, A-fragment friendly) ----
    if constexpr (LAYER == 0) {
      for (int p = tid; p < MTILE * 32; p += 256) {
        int row = p >> 5, j = p & 31;
        float v = (j < INPUT) ? x[(size_t)(b0 + row) * (SEQ * INPUT) + t * INPUT + j] : 0.0f;
        sX[p] = f2bf(v);
      }
    } else {
      const uint4* src = (const uint4*)h0seq;
      uint4* dst = (uint4*)sX;
      for (int p = tid; p < 1024; p += 256) {
        int row = p >> 4, c8 = p & 15;
        dst[p] = src[((size_t)(b0 + row) * SEQ + t) * 16 + c8];
      }
    }
    __syncthreads();

    // ---- input projection: x_t @ W_ih^T  (first WMMA consumes bias as C) ----
#pragma unroll
    for (int ks = 0; ks < KSX; ++ks) {
      v16bf Bf[4];
#pragma unroll
      for (int g = 0; g < 4; ++g)
        Bf[g] = loadB(sWih + (size_t)(ks * 32 + wave * 4 + g) * 512, lane);
#pragma unroll
      for (int mt = 0; mt < 4; ++mt) {
        v16bf a = loadA(sX, XSTRIDE, mt * 16, ks * 32, lane);
#pragma unroll
        for (int g = 0; g < 4; ++g)
          acc[mt][g] = __builtin_amdgcn_wmma_f32_16x16x32_bf16(
              false, a, false, Bf[g], (short)0,
              (ks == 0) ? biasv[g] : acc[mt][g], false, false);
      }
    }
    // ---- recurrent projection: h_{t-1} @ W_hh^T ----
#pragma unroll
    for (int ks = 0; ks < 4; ++ks) {
      v16bf Bf[4];
#pragma unroll
      for (int g = 0; g < 4; ++g)
        Bf[g] = loadB(sWhh + (size_t)(ks * 32 + wave * 4 + g) * 512, lane);
#pragma unroll
      for (int mt = 0; mt < 4; ++mt) {
        v16bf a = loadA(sH, HIDDEN, mt * 16, ks * 32, lane);
#pragma unroll
        for (int g = 0; g < 4; ++g)
          acc[mt][g] = __builtin_amdgcn_wmma_f32_16x16x32_bf16(
              false, a, false, Bf[g], (short)0, acc[mt][g], false, false);
      }
    }
    __syncthreads();   // everyone done reading sH before overwrite

    // ---- LSTM cell (entirely wave-local: i,f,g,o and c in VGPRs) ----
    {
      const int col = wave * 16 + nloc;
#pragma unroll
      for (int mt = 0; mt < 4; ++mt) {
#pragma unroll
        for (int e = 0; e < 8; ++e) {
          float iv = acc[mt][0][e], fv = acc[mt][1][e];
          float gv = acc[mt][2][e], ov = acc[mt][3][e];
          float cn = sigf(fv) * cst[mt][e] + sigf(iv) * tanhfast(gv);
          float hn = sigf(ov) * tanhfast(cn);
          cst[mt][e] = cn;
          sH[(mt * 16 + hiB * 8 + e) * HIDDEN + col] = f2bf(hn);
        }
      }
    }
    __syncthreads();

    if constexpr (LAYER == 0) {   // publish h0_t for layer 1 (L2-resident)
      const uint4* srch = (const uint4*)sH;
      uint4* dst = (uint4*)h0seq;
      for (int p = tid; p < 1024; p += 256) {
        int row = p >> 4, c8 = p & 15;
        dst[((size_t)(b0 + row) * SEQ + t) * 16 + c8] = srch[p];
      }
    }
  }

  // ---- epilogue: h_n, c_n (+ logits for layer 1) ----
  const size_t HB = (size_t)BATCH * HIDDEN;
  float* hout = out + (size_t)BATCH * OUTD + (size_t)LAYER * HB;
  float* cout = out + (size_t)BATCH * OUTD + 2 * HB + (size_t)LAYER * HB;
  {
    const int col = wave * 16 + nloc;
#pragma unroll
    for (int mt = 0; mt < 4; ++mt)
#pragma unroll
      for (int e = 0; e < 8; ++e) {
        int row = b0 + mt * 16 + hiB * 8 + e;
        cout[(size_t)row * HIDDEN + col] = cst[mt][e];
      }
  }
  for (int p = tid; p < MTILE * HIDDEN; p += 256) {
    int row = p >> 7, c = p & 127;
    hout[(size_t)(b0 + row) * HIDDEN + c] = bf2f(sH[p]);
  }
  if constexpr (LAYER == 1) {
    for (int p = tid; p < MTILE * OUTD; p += 256) {
      int row = p / OUTD, o = p % OUTD;
      float s = fc_b[o];
#pragma unroll 4
      for (int k = 0; k < HIDDEN; ++k)
        s += bf2f(sH[row * HIDDEN + k]) * fc_w[o * HIDDEN + k];
      out[(size_t)(b0 + row) * OUTD + o] = s;
    }
  }
}

extern "C" void kernel_launch(void* const* d_in, const int* in_sizes, int n_in,
                              void* d_out, int out_size, void* d_ws, size_t ws_size,
                              hipStream_t stream) {
  (void)in_sizes; (void)n_in; (void)out_size; (void)ws_size;
  const float* x    = (const float*)d_in[0];
  const float* wih0 = (const float*)d_in[1];
  const float* whh0 = (const float*)d_in[2];
  const float* bih0 = (const float*)d_in[3];
  const float* bhh0 = (const float*)d_in[4];
  const float* wih1 = (const float*)d_in[5];
  const float* whh1 = (const float*)d_in[6];
  const float* bih1 = (const float*)d_in[7];
  const float* bhh1 = (const float*)d_in[8];
  const float* fcw  = (const float*)d_in[9];
  const float* fcb  = (const float*)d_in[10];
  float* out = (float*)d_out;

  u16*   wblob = (u16*)d_ws;
  float* bias0 = (float*)((char*)d_ws + BIAS0_BYTE);
  float* bias1 = (float*)((char*)d_ws + BIAS1_BYTE);
  u16*   h0seq = (u16*)((char*)d_ws + H0SEQ_BYTE);

  prep_kernel<<<836, 256, 0, stream>>>(wih0, whh0, bih0, bhh0,
                                       wih1, whh1, bih1, bhh1,
                                       wblob, bias0, bias1);

  lstm_layer_kernel<0><<<NBLK, 256, 0, stream>>>(
      x, wblob + 0, wblob + 16384, bias0, h0seq, nullptr, nullptr, out);

  lstm_layer_kernel<1><<<NBLK, 256, 0, stream>>>(
      nullptr, wblob + 81920, wblob + 147456, bias1, h0seq, fcw, fcb, out);
}
